// NeRF_15693810500008
// MI455X (gfx1250) — matrix operand order repaired and checked by reading, not compiled, not run on hardware
//
#include <hip/hip_runtime.h>

// ---------------------------------------------------------------------------
// Instant-NGP NeRF forward for MI455X (gfx1250, wave32, WMMA).
//  - hash encoding: per-lane gathers, restructured as gather-then-accumulate
//    over level pairs so 16 global_load_b64 are in flight before any wait.
//  - MLP: V_WMMA_F32_16X16X32_F16, weights stationary in LDS (f16, transposed),
//    activations round-trip through per-wave LDS scratch between layers.
// ---------------------------------------------------------------------------

typedef __attribute__((ext_vector_type(16))) _Float16 h16_t;
typedef __attribute__((ext_vector_type(8)))  _Float16 h8_t;
typedef __attribute__((ext_vector_type(2)))  _Float16 h2_t;
typedef __attribute__((ext_vector_type(8)))  float    f8_t;

#define BATCH     524288
#define NLEV      16
#define NTHR      128                  // 4 wave32
#define WAVES     4
#define PTS_BLK   128
#define NCHUNK    (BATCH / PTS_BLK)    // 4096
#define NBLK      2048

// LDS strides (in halves): multiples of 8 halves (16B, b128 alignment),
// 80B/144B = 20/36 banks -> 16 lanes x 16B touch all 64 banks once.
#define STR32     40
#define STR64     72
#define SCR       (32 * STR64)         // 2304 halves per-wave scratch

// weight LDS offsets (halves); all offsets*2 are 16B-aligned
#define OFF_D1    0
#define OFF_D2    (OFF_D1 + 64 * STR32)   // 2560
#define OFF_D3    (OFF_D2 + 64 * STR64)   // 7168
#define OFF_C1    (OFF_D3 + 16 * STR64)   // 8320
#define OFF_C2    (OFF_C1 + 64 * STR32)   // 10880
#define OFF_C3    (OFF_C2 + 64 * STR64)   // 15488
#define OFF_C4    (OFF_C3 + 64 * STR64)   // 20096
#define W_TOTAL   (OFF_C4 + 16 * STR64)   // 21248 halves = 42496 B

// bias LDS offsets (floats); all multiples of 8 (32B v8f loads)
#define BOFF_D1   0
#define BOFF_D2   64
#define BOFF_D3   128
#define BOFF_C1   144
#define BOFF_C2   208
#define BOFF_C3   272
#define BOFF_C4   336
#define B_TOTAL   352

#define FEAT_SCALE 256.0f

// --- fragment helpers ------------------------------------------------------

__device__ __forceinline__ h16_t frag_cat(h8_t a, h8_t b) {
  return __builtin_shufflevector(a, b, 0,1,2,3,4,5,6,7,8,9,10,11,12,13,14,15);
}

// A-matrix 16x32 f16 layout: lanes 0-15 row M, halves K{0-7,16-23};
// lanes 16-31 row M, halves K{8-15,24-31}.
__device__ __forceinline__ h16_t lds_a(const _Float16* base, int row, int stride,
                                       int kbase, int hi) {
  const _Float16* p = base + row * stride + kbase + (hi << 3);
  return frag_cat(*(const h8_t*)p, *(const h8_t*)(p + 16));
}

// B-matrix 32x16 f16 layout: lane holds column N, halves K{0-7,16-23} (lanes 0-15)
// or K{8-15,24-31} (lanes 16-31), matching the A layout.
__device__ __forceinline__ h16_t lds_b(const _Float16* base, int col, int stride,
                                       int kbase, int hi) {
  const _Float16* p = base + col * stride + kbase + (hi << 3);
  return frag_cat(*(const h8_t*)p, *(const h8_t*)(p + 16));
}

__device__ __forceinline__ f8_t wmma32(h16_t a, h16_t b, f8_t c) {
  return __builtin_amdgcn_wmma_f32_16x16x32_f16(false, a, false, b,
                                                (short)0, c, false, false);
}

__device__ __forceinline__ f8_t bias_act(f8_t d, const float* bp, bool relu) {
  f8_t b = *(const f8_t*)bp;
#pragma unroll
  for (int i = 0; i < 8; ++i) {
    float v = d[i] + b[i];
    d[i] = relu ? fmaxf(v, 0.0f) : v;
  }
  return d;
}

__device__ __forceinline__ void store8h(_Float16* p, f8_t d) {
  h8_t h;
#pragma unroll
  for (int i = 0; i < 8; ++i) h[i] = (_Float16)d[i];
  *(h8_t*)p = h;
}

__device__ __forceinline__ float sigm(float v) {
  return 1.0f / (1.0f + __expf(-v));
}

// --- layers ----------------------------------------------------------------

// 64 -> 64 with ReLU; act read+written in scr (stride 72), two 16-pt tiles.
__device__ __forceinline__ void hidden64(_Float16* scr, const _Float16* wt,
                                         const float* bias, int l15, int hi) {
  h16_t b00 = lds_b(scr, l15,      STR64, 0,  hi);
  h16_t b01 = lds_b(scr, l15,      STR64, 32, hi);
  h16_t b10 = lds_b(scr, 16 + l15, STR64, 0,  hi);
  h16_t b11 = lds_b(scr, 16 + l15, STR64, 32, hi);
#pragma unroll
  for (int c = 0; c < 4; ++c) {
    h16_t a0 = lds_a(wt, c * 16 + l15, STR64, 0,  hi);
    h16_t a1 = lds_a(wt, c * 16 + l15, STR64, 32, hi);
    f8_t z = {};
    f8_t d0 = wmma32(a1, b01, wmma32(a0, b00, z));
    f8_t d1 = wmma32(a1, b11, wmma32(a0, b10, z));
    const float* bp = bias + c * 16 + hi * 8;
    d0 = bias_act(d0, bp, true);
    d1 = bias_act(d1, bp, true);
    store8h(scr + l15 * STR64        + c * 16 + hi * 8, d0);
    store8h(scr + (16 + l15) * STR64 + c * 16 + hi * 8, d1);
  }
}

// 64 -> 16 (no ReLU); results kept in registers (lane=point, vgpr=feature).
__device__ __forceinline__ void out16(_Float16* scr, const _Float16* wt,
                                      const float* bias, int l15, int hi,
                                      f8_t& r0, f8_t& r1) {
  h16_t b00 = lds_b(scr, l15,      STR64, 0,  hi);
  h16_t b01 = lds_b(scr, l15,      STR64, 32, hi);
  h16_t b10 = lds_b(scr, 16 + l15, STR64, 0,  hi);
  h16_t b11 = lds_b(scr, 16 + l15, STR64, 32, hi);
  h16_t a0 = lds_a(wt, l15, STR64, 0,  hi);
  h16_t a1 = lds_a(wt, l15, STR64, 32, hi);
  f8_t z = {};
  const float* bp = bias + hi * 8;
  r0 = bias_act(wmma32(a1, b01, wmma32(a0, b00, z)), bp, false);
  r1 = bias_act(wmma32(a1, b11, wmma32(a0, b10, z)), bp, false);
}

// 32 -> 64 with ReLU; B from scr with stride bstride (feat=40 or act=72).
__device__ __forceinline__ void in32x64(_Float16* scr, int bstride,
                                        const _Float16* wt, const float* bias,
                                        int l15, int hi) {
  h16_t b0 = lds_b(scr, l15,      bstride, 0, hi);
  h16_t b1 = lds_b(scr, 16 + l15, bstride, 0, hi);
#pragma unroll
  for (int c = 0; c < 4; ++c) {
    h16_t a = lds_a(wt, c * 16 + l15, STR32, 0, hi);
    f8_t z = {};
    f8_t d0 = wmma32(a, b0, z);
    f8_t d1 = wmma32(a, b1, z);
    const float* bp = bias + c * 16 + hi * 8;
    d0 = bias_act(d0, bp, true);
    d1 = bias_act(d1, bp, true);
    store8h(scr + l15 * STR64        + c * 16 + hi * 8, d0);
    store8h(scr + (16 + l15) * STR64 + c * 16 + hi * 8, d1);
  }
}

// --- hash encoding ---------------------------------------------------------

struct LevelQ {
  int   i0, i1, i2;
  float w0, w1, w2;
};

__device__ __forceinline__ LevelQ level_setup(int res, float x0, float x1, float x2) {
  LevelQ q;
  const float s = (float)(res - 1);
  float q0 = x0 * s, q1 = x1 * s, q2 = x2 * s;
  q.i0 = (int)floorf(q0); q.i0 = q.i0 < 0 ? 0 : (q.i0 > res - 2 ? res - 2 : q.i0);
  q.i1 = (int)floorf(q1); q.i1 = q.i1 < 0 ? 0 : (q.i1 > res - 2 ? res - 2 : q.i1);
  q.i2 = (int)floorf(q2); q.i2 = q.i2 < 0 ? 0 : (q.i2 > res - 2 ? res - 2 : q.i2);
  q.w0 = q0 - (float)q.i0; q.w1 = q1 - (float)q.i1; q.w2 = q2 - (float)q.i2;
  return q;
}

__device__ __forceinline__ unsigned level_index(int res, const LevelQ& q, int k) {
  const int ox = (k >> 2) & 1, oy = (k >> 1) & 1, oz = k & 1;
  const int cx = q.i0 + ox, cy = q.i1 + oy, cz = q.i2 + oz;
  if (res * res * res <= 65536) {            // dense levels 0..4
    return (unsigned)(cx + res * (cy + res * cz));
  }
  return ((unsigned)cx ^ (unsigned)cy * 2654435761u
                       ^ (unsigned)cz * 805459861u) & 65535u;   // hash mod 2^16
}

__device__ __forceinline__ void level_accum(const LevelQ& q, const float2* e,
                                            float& f0, float& f1) {
#pragma unroll
  for (int k = 0; k < 8; ++k) {
    const int ox = (k >> 2) & 1, oy = (k >> 1) & 1, oz = k & 1;
    float wc = (ox ? q.w0 : 1.0f - q.w0) * (oy ? q.w1 : 1.0f - q.w1)
             * (oz ? q.w2 : 1.0f - q.w2);
    f0 = fmaf(wc, e[k].x, f0);
    f1 = fmaf(wc, e[k].y, f1);
  }
}

__device__ __forceinline__ void encode_point(const float* xp, const float* tbl,
                                             _Float16* featRow) {
  float x0 = xp[0] * 0.5f + 0.5f;
  float x1 = xp[1] * 0.5f + 0.5f;
  float x2 = xp[2] * 0.5f + 0.5f;
  constexpr int RESL[NLEV] = {16, 20, 25, 32, 40, 50, 64, 80,
                              101, 128, 161, 203, 256, 322, 406, 512};
  // Process two levels per step: issue all 16 gathers, then accumulate.
#pragma unroll
  for (int lp = 0; lp < NLEV; lp += 2) {
    const int resA = RESL[lp], resB = RESL[lp + 1];
    const LevelQ qa = level_setup(resA, x0, x1, x2);
    const LevelQ qb = level_setup(resB, x0, x1, x2);
    const float* tA = tbl + lp * 131072;        // 65536*2 floats per level
    const float* tB = tA + 131072;

    float2 eA[8], eB[8];
#pragma unroll
    for (int k = 0; k < 8; ++k)
      eA[k] = *(const float2*)(tA + 2 * level_index(resA, qa, k));
#pragma unroll
    for (int k = 0; k < 8; ++k)
      eB[k] = *(const float2*)(tB + 2 * level_index(resB, qb, k));

    float fa0 = 0.0f, fa1 = 0.0f, fb0 = 0.0f, fb1 = 0.0f;
    level_accum(qa, eA, fa0, fa1);
    level_accum(qb, eB, fb0, fb1);

    h2_t ha; ha.x = (_Float16)(fa0 * FEAT_SCALE); ha.y = (_Float16)(fa1 * FEAT_SCALE);
    h2_t hb; hb.x = (_Float16)(fb0 * FEAT_SCALE); hb.y = (_Float16)(fb1 * FEAT_SCALE);
    *(h2_t*)(featRow + 2 * lp)     = ha;
    *(h2_t*)(featRow + 2 * lp + 2) = hb;
  }
}

// --- weight staging --------------------------------------------------------

// dst[n*STRIDE + k] = (k<kReal && n<srcN) ? src[k*srcN + n]*scale : 0
template <int K, int STRIDE>
__device__ __forceinline__ void fill_wt(_Float16* dst, const float* src, int N,
                                        int kReal, int srcN, float scale, int tid) {
  for (int i = tid; i < N * K; i += NTHR) {
    int n = i / K, k = i - n * K;
    float v = (k < kReal && n < srcN) ? src[k * srcN + n] * scale : 0.0f;
    dst[n * STRIDE + k] = (_Float16)v;
  }
}

// --- kernel ----------------------------------------------------------------

__global__ __launch_bounds__(NTHR)
void ngp_forward(const float* __restrict__ x, const float* __restrict__ tables,
                 const float* __restrict__ dw1, const float* __restrict__ db1,
                 const float* __restrict__ dw2, const float* __restrict__ db2,
                 const float* __restrict__ dw3, const float* __restrict__ db3,
                 const float* __restrict__ cw1, const float* __restrict__ cb1,
                 const float* __restrict__ cw2, const float* __restrict__ cb2,
                 const float* __restrict__ cw3, const float* __restrict__ cb3,
                 const float* __restrict__ cw4, const float* __restrict__ cb4,
                 float* __restrict__ out) {
  __shared__ __align__(16) _Float16 s_w[W_TOTAL];      // 42496 B
  __shared__ __align__(32) float    s_bias[B_TOTAL];   //  1408 B
  __shared__ __align__(16) _Float16 s_scr[WAVES * SCR];// 18432 B  -> 62336 B total

  const int tid  = threadIdx.x;
  const int wave = tid >> 5;
  const int lane = tid & 31;
  const int l15  = lane & 15;
  const int hi   = lane >> 4;

  // stage weights (f16, transposed Wt[n][k]) and biases into LDS
  fill_wt<32, STR32>(s_w + OFF_D1, dw1, 64, 32, 64, 1.0f / FEAT_SCALE, tid);
  fill_wt<64, STR64>(s_w + OFF_D2, dw2, 64, 64, 64, 1.0f, tid);
  fill_wt<64, STR64>(s_w + OFF_D3, dw3, 16, 64, 16, 1.0f, tid);
  fill_wt<32, STR32>(s_w + OFF_C1, cw1, 64, 16, 64, 1.0f, tid);  // K pad 16..31 = 0
  fill_wt<64, STR64>(s_w + OFF_C2, cw2, 64, 64, 64, 1.0f, tid);
  fill_wt<64, STR64>(s_w + OFF_C3, cw3, 64, 64, 64, 1.0f, tid);
  fill_wt<64, STR64>(s_w + OFF_C4, cw4, 16, 64, 3,  1.0f, tid);  // rows 3..15 = 0
  for (int i = tid; i < B_TOTAL; i += NTHR) {
    float v;
    if      (i < 64)  v = db1[i];
    else if (i < 128) v = db2[i - 64];
    else if (i < 144) v = db3[i - 128];
    else if (i < 208) v = cb1[i - 144];
    else if (i < 272) v = cb2[i - 208];
    else if (i < 336) v = cb3[i - 272];
    else { int j = i - 336; v = (j < 3) ? cb4[j] : 0.0f; }
    s_bias[i] = v;
  }
  __syncthreads();

  _Float16* scr = s_scr + wave * SCR;

  for (int ch = blockIdx.x; ch < NCHUNK; ch += NBLK) {
    const int pbase = ch * PTS_BLK + wave * 32;

    // 1) hash encoding: lane -> point pbase+lane, 32 f16 feats (x256) into scr
    encode_point(x + 3 * (pbase + lane), tables, scr + lane * STR32);
    // same-wave DS ops are in-order: feat stores complete before B loads below.

    // 2) density MLP
    in32x64(scr, STR32, s_w + OFF_D1, s_bias + BOFF_D1, l15, hi);   // 32->64 relu
    hidden64(scr, s_w + OFF_D2, s_bias + BOFF_D2, l15, hi);         // 64->64 relu
    f8_t s0, s1;
    out16(scr, s_w + OFF_D3, s_bias + BOFF_D3, l15, hi, s0, s1);    // 64->16

    if (lane < 16) {                       // feature 0 lives on lanes 0-15, vgpr 0
      out[pbase + lane]      = __expf(s0[0]);
      out[pbase + 16 + lane] = __expf(s1[0]);
    }

    // write d (16 feats) into act rows, zero-padding K 16..31 for the c1 layer
    {
      h8_t zz = {};
      store8h(scr + l15 * STR64 + hi * 8, s0);
      *(h8_t*)(scr + l15 * STR64 + 16 + hi * 8) = zz;
      store8h(scr + (16 + l15) * STR64 + hi * 8, s1);
      *(h8_t*)(scr + (16 + l15) * STR64 + 16 + hi * 8) = zz;
    }

    // 3) color MLP
    in32x64(scr, STR64, s_w + OFF_C1, s_bias + BOFF_C1, l15, hi);   // 16(pad32)->64
    hidden64(scr, s_w + OFF_C2, s_bias + BOFF_C2, l15, hi);
    hidden64(scr, s_w + OFF_C3, s_bias + BOFF_C3, l15, hi);
    f8_t c0v, c1v;
    out16(scr, s_w + OFF_C4, s_bias + BOFF_C4, l15, hi, c0v, c1v);  // 64->3(pad16)

    if (lane < 16) {
      float* o0 = out + BATCH + 3 * (pbase + lane);
      o0[0] = sigm(c0v[0]); o0[1] = sigm(c0v[1]); o0[2] = sigm(c0v[2]);
      float* o1 = out + BATCH + 3 * (pbase + 16 + lane);
      o1[0] = sigm(c1v[0]); o1[1] = sigm(c1v[1]); o1[2] = sigm(c1v[2]);
    }
  }
}

// --- host-side launch ------------------------------------------------------

extern "C" void kernel_launch(void* const* d_in, const int* in_sizes, int n_in,
                              void* d_out, int out_size, void* d_ws, size_t ws_size,
                              hipStream_t stream) {
  (void)in_sizes; (void)n_in; (void)out_size; (void)d_ws; (void)ws_size;
  ngp_forward<<<NBLK, NTHR, 0, stream>>>(
      (const float*)d_in[0],  (const float*)d_in[1],
      (const float*)d_in[2],  (const float*)d_in[3],
      (const float*)d_in[4],  (const float*)d_in[5],
      (const float*)d_in[6],  (const float*)d_in[7],
      (const float*)d_in[8],  (const float*)d_in[9],
      (const float*)d_in[10], (const float*)d_in[11],
      (const float*)d_in[12], (const float*)d_in[13],
      (const float*)d_in[14], (const float*)d_in[15],
      (float*)d_out);
}
